// MultiHeadAttention_9165460210007
// MI455X (gfx1250) — compile-verified, matrix operand order checked by reading, not run on hardware
//
#include <hip/hip_runtime.h>
#include <hip/hip_bf16.h>
#include <math.h>

// Problem constants
#define BB   2
#define SS   2048
#define EE   1024
#define HH   16
#define DK   64

// Toggle for the CDNA5 async global->LDS copy path (inline asm).
#define USE_ASYNC_LDS 1

typedef __attribute__((ext_vector_type(16))) _Float16 v16h;
typedef __attribute__((ext_vector_type(8)))  _Float16 v8h;
typedef __attribute__((ext_vector_type(4)))  _Float16 v4h;
typedef __attribute__((ext_vector_type(8)))  float    v8f;
typedef __attribute__((ext_vector_type(4)))  float    v4f;
typedef __attribute__((ext_vector_type(4)))  int      v4i;

// ---------------------------------------------------------------------------
// WMMA helpers
// ---------------------------------------------------------------------------
__device__ __forceinline__ v8f wmma_f16(v16h a, v16h b, v8f c) {
    return __builtin_amdgcn_wmma_f32_16x16x32_f16(
        /*neg_a=*/false, a, /*neg_b=*/false, b,
        /*c_mod=*/(short)0, c, /*reuse_a=*/false, /*reuse_b=*/false);
}

// 16x32 f16 A-fragment (B mirrors it with lane = column).  CDNA5 ISA 7.12.2:
//   lanes 0-15 : row = lane,    VGPR0-3 = K 0..7,  VGPR4-7 = K 16..23
//   lanes16-31 : row = lane-16, VGPR0-3 = K 8..15, VGPR4-7 = K 24..31
// => per lane two contiguous 16B reads at k0 and k0+16.
__device__ __forceinline__ v16h frag_from_ptr(const _Float16* p) {
    v8h lo = *(const v8h*)(p);
    v8h hi = *(const v8h*)(p + 16);
    v16h f;
#pragma unroll
    for (int i = 0; i < 8; ++i) { f[i] = lo[i]; f[i + 8] = hi[i]; }
    return f;
}

__device__ __forceinline__ v16h frag_from_tile(const _Float16* base, int stride) {
    int lane = threadIdx.x & 31;
    int row  = lane & 15;
    int k0   = (lane & 16) ? 8 : 0;
    return frag_from_ptr(base + row * stride + k0);
}

#if USE_ASYNC_LDS
// Async copy 16B per lane: LDS[ldsaddr] = MEM[gaddr].  Tracked by ASYNCcnt.
// Generic LDS pointer low 32 bits are the LDS byte offset (ISA 10.2 aperture).
__device__ __forceinline__ void async_copy_b128(const void* gaddr, void* ldsaddr) {
    unsigned l = (unsigned)(uintptr_t)ldsaddr;
    asm volatile("global_load_async_to_lds_b128 %0, %1, off"
                 :: "v"(l), "v"(gaddr) : "memory");
}
__device__ __forceinline__ void wait_asynccnt0() {
    asm volatile("s_wait_asynccnt 0x0" ::: "memory");
}
#endif

// ---------------------------------------------------------------------------
// 1) f32 -> f16 convert
// ---------------------------------------------------------------------------
__global__ void cvt_f32_to_f16_kernel(const float* __restrict__ src,
                                      _Float16* __restrict__ dst, int n4) {
    int i = blockIdx.x * blockDim.x + threadIdx.x;
    if (i < n4) {
        v4f t = *(const v4f*)(src + (size_t)i * 4);
        v4h o;
#pragma unroll
        for (int j = 0; j < 4; ++j) o[j] = (_Float16)t[j];
        *(v4h*)(dst + (size_t)i * 4) = o;
    }
}

// ---------------------------------------------------------------------------
// 2/6) WMMA GEMM: C[M,N] = A[M,K](f16) @ B[K,N](f16) + bias
//   MODE 0: store f16 head-split [B,H,S,DK]   (QKV projections)
//   MODE 1: store f32 row-major  [M,N]         (final output)
// 256 threads (8 waves), tile 128(M) x 64(N), K staged 32 at a time.
// ---------------------------------------------------------------------------
template <int MODE>
__global__ __launch_bounds__(256)
void gemm_kernel(const _Float16* __restrict__ A, const _Float16* __restrict__ Bm,
                 const float* __restrict__ bias, void* __restrict__ Out,
                 int M, int N, int K) {
    __shared__ __align__(64) _Float16 lds_a[128 * 48];   // A tile [128][32], stride 48
    __shared__ __align__(64) _Float16 lds_bt[64 * 48];   // B tile transposed [64 cols][32 k]

    const int tid = threadIdx.x;
    const int wave = tid >> 5;          // 0..7
    const int lane = tid & 31;
    const int m0 = blockIdx.y * 128;
    const int n0 = blockIdx.x * 64;

    const int arow = tid >> 1, aseg = tid & 1;   // A: 16 halves / thread
    const int krow = tid >> 3, bseg = tid & 7;   // B: 8 halves / thread

    v8f acc[4] = {v8f{}, v8f{}, v8f{}, v8f{}};

    for (int kt = 0; kt < K; kt += 32) {
        const _Float16* ag = A + (size_t)(m0 + arow) * K + kt + aseg * 16;
        _Float16* al = lds_a + arow * 48 + aseg * 16;
#if USE_ASYNC_LDS
        async_copy_b128(ag, al);
        async_copy_b128(ag + 8, al + 8);
#else
        *(v16h*)al = *(const v16h*)ag;
#endif
        {   // stage B transposed: read row fragments, scatter as columns
            v8h bv = *(const v8h*)(Bm + (size_t)(kt + krow) * N + n0 + bseg * 8);
#pragma unroll
            for (int j = 0; j < 8; ++j)
                lds_bt[(bseg * 8 + j) * 48 + krow] = bv[j];
        }
#if USE_ASYNC_LDS
        wait_asynccnt0();
#endif
        __syncthreads();

        v16h afrag = frag_from_tile(lds_a + wave * 16 * 48, 48);
#pragma unroll
        for (int nt = 0; nt < 4; ++nt) {
            v16h bfrag = frag_from_tile(lds_bt + nt * 16 * 48, 48);
            acc[nt] = wmma_f16(afrag, bfrag, acc[nt]);
        }
        __syncthreads();
    }

    // C layout: lanes 0-15 hold M = vgpr, N = lane; lanes 16-31 M = 8+vgpr.
    const int nl = lane & 15;
    const int mh = (lane >> 4) * 8;
#pragma unroll
    for (int nt = 0; nt < 4; ++nt) {
        int n = n0 + nt * 16 + nl;
        float bv = bias[n];
#pragma unroll
        for (int r = 0; r < 8; ++r) {
            int m = m0 + wave * 16 + mh + r;
            float val = acc[nt][r] + bv;
            if (MODE == 0) {
                int h = n >> 6, d = n & (DK - 1);
                int b = m >> 11, s = m & (SS - 1);
                ((_Float16*)Out)[((((size_t)b * HH + h) * SS + s) << 6) + d] =
                    (_Float16)val;
            } else {
                ((float*)Out)[(size_t)m * N + n] = val;
            }
        }
    }
}

// ---------------------------------------------------------------------------
// 3) Scores: raw S = scale * Q Kᵀ streamed to weights buffer.
//    Per-lane online (max, sum-exp) over the columns each lane owns
//    (col mod 16 == lane&15); one pairwise cross-lane merge at the end.
// ---------------------------------------------------------------------------
__global__ __launch_bounds__(32)
void scores_kernel(const _Float16* __restrict__ q, const _Float16* __restrict__ k,
                   float* __restrict__ wout, float* __restrict__ stats) {
    const int lane = threadIdx.x & 31;
    const int bh = blockIdx.y;            // b*H + h
    const int q0 = blockIdx.x * 16;
    const int row = lane & 15;
    const int k0 = (lane & 16) ? 8 : 0;

    const _Float16* qb = q + ((size_t)bh * SS + q0) * DK + row * DK + k0;
    v16h aq0 = frag_from_ptr(qb);        // K = 0..31 of DK
    v16h aq1 = frag_from_ptr(qb + 32);   // K = 32..63

    float mrun[8], lrun[8];
#pragma unroll
    for (int r = 0; r < 8; ++r) { mrun[r] = -INFINITY; lrun[r] = 0.0f; }

    const float scale = 0.125f;          // 1/sqrt(64)
    const int mh = (lane >> 4) * 8;
    const int nl = lane & 15;

    for (int kt = 0; kt < SS; kt += 16) {
        const _Float16* kb =
            k + ((size_t)bh * SS + kt) * DK + row * DK + k0;   // lane = key column
        v16h b0 = frag_from_ptr(kb);
        v16h b1 = frag_from_ptr(kb + 32);
        v8f c = v8f{};
        c = wmma_f16(aq0, b0, c);
        c = wmma_f16(aq1, b1, c);

#pragma unroll
        for (int r = 0; r < 8; ++r) {
            float s = c[r] * scale;
            float mo = mrun[r];
            float mn = fmaxf(mo, s);
            lrun[r] = lrun[r] * __expf(mo - mn) + __expf(s - mn);
            mrun[r] = mn;
            wout[((size_t)bh * SS + q0 + mh + r) * SS + kt + nl] = s;
        }
    }

    // merge (m,l) pairs across the 16 lanes holding each row
#pragma unroll
    for (int r = 0; r < 8; ++r) {
        float m = mrun[r], l = lrun[r];
#pragma unroll
        for (int msk = 1; msk < 16; msk <<= 1) {
            float mo = __shfl_xor(m, msk, 32);
            float lo = __shfl_xor(l, msk, 32);
            float mn = fmaxf(m, mo);
            l = l * __expf(m - mn) + lo * __expf(mo - mn);
            m = mn;
        }
        if ((lane & 15) == 0) {
            size_t idx = ((size_t)bh * SS + q0 + mh + r) * 2;
            stats[idx] = m;
            stats[idx + 1] = l;
        }
    }
}

// ---------------------------------------------------------------------------
// 4) Normalize + post-softmax mask zeroing (no renorm), in place on weights.
// ---------------------------------------------------------------------------
__global__ void norm_mask_kernel(float* __restrict__ w,
                                 const float* __restrict__ stats,
                                 const int* __restrict__ mask) {
    size_t i = ((size_t)blockIdx.x * blockDim.x + threadIdx.x) * 4;
    size_t rowg = i >> 11;               // / SS
    int col = (int)(i & (SS - 1));
    int qs = (int)(rowg & (SS - 1));
    float m = stats[rowg * 2];
    float inv = 1.0f / stats[rowg * 2 + 1];
    v4f s = *(v4f*)(w + i);
    v4i mk = *(const v4i*)(mask + (size_t)qs * SS + col);
    v4f o;
#pragma unroll
    for (int j = 0; j < 4; ++j)
        o[j] = mk[j] ? __expf(s[j] - m) * inv : 0.0f;
    *(v4f*)(w + i) = o;
}

// ---------------------------------------------------------------------------
// 5) attn = weights(f32 -> f16 on the fly) @ V(f16)
//    256 threads (8 waves), tile 128(M) x 64(N=DK), K over S in steps of 32.
// ---------------------------------------------------------------------------
__global__ __launch_bounds__(256)
void attn_kernel(const float* __restrict__ w, const _Float16* __restrict__ v,
                 _Float16* __restrict__ attn) {
    __shared__ __align__(64) _Float16 lds_a[128 * 48];
    __shared__ __align__(64) _Float16 lds_bt[64 * 48];

    const int tid = threadIdx.x;
    const int wave = tid >> 5;
    const int lane = tid & 31;
    const int bh = blockIdx.y;
    const int b = bh >> 4, h = bh & (HH - 1);
    const int q0 = blockIdx.x * 128;

    const float* wbase = w + ((size_t)bh * SS + q0) * SS;
    const _Float16* vbase = v + (size_t)bh * SS * DK;

    const int arow = tid >> 1, aseg = tid & 1;   // 16 floats / thread
    const int krow = tid >> 3, bseg = tid & 7;   // 8 halves / thread

    v8f acc[4] = {v8f{}, v8f{}, v8f{}, v8f{}};

    for (int kt = 0; kt < SS; kt += 32) {
        {   // stage weights tile 128x32 f32 -> f16
            const float* p = wbase + (size_t)arow * SS + kt + aseg * 16;
            v16h av;
#pragma unroll
            for (int j = 0; j < 16; j += 4) {
                v4f t = *(const v4f*)(p + j);
                av[j] = (_Float16)t[0]; av[j + 1] = (_Float16)t[1];
                av[j + 2] = (_Float16)t[2]; av[j + 3] = (_Float16)t[3];
            }
            *(v16h*)(lds_a + arow * 48 + aseg * 16) = av;
        }
        {   // stage V tile [32 k][64 d] transposed
            v8h bv = *(const v8h*)(vbase + (size_t)(kt + krow) * DK + bseg * 8);
#pragma unroll
            for (int j = 0; j < 8; ++j)
                lds_bt[(bseg * 8 + j) * 48 + krow] = bv[j];
        }
        __syncthreads();

        v16h afrag = frag_from_tile(lds_a + wave * 16 * 48, 48);
#pragma unroll
        for (int nt = 0; nt < 4; ++nt) {
            v16h bfrag = frag_from_tile(lds_bt + nt * 16 * 48, 48);
            acc[nt] = wmma_f16(afrag, bfrag, acc[nt]);
        }
        __syncthreads();
    }

    // store to [B, S, H, DK] f16 (== [B,S,E] flattened for the final GEMM)
    const int nl = lane & 15;
    const int mh = (lane >> 4) * 8;
#pragma unroll
    for (int nt = 0; nt < 4; ++nt) {
        int d = nt * 16 + nl;
#pragma unroll
        for (int r = 0; r < 8; ++r) {
            int s = q0 + wave * 16 + mh + r;
            attn[(((size_t)b * SS + s) * HH + h) * DK + d] = (_Float16)acc[nt][r];
        }
    }
}

// ---------------------------------------------------------------------------
// Launch
// ---------------------------------------------------------------------------
extern "C" void kernel_launch(void* const* d_in, const int* in_sizes, int n_in,
                              void* d_out, int out_size, void* d_ws, size_t ws_size,
                              hipStream_t stream) {
    const float* x    = (const float*)d_in[0];
    const int*   mask = (const int*)d_in[1];
    const float* Wq = (const float*)d_in[2]; const float* bq = (const float*)d_in[3];
    const float* Wk = (const float*)d_in[4]; const float* bk = (const float*)d_in[5];
    const float* Wv = (const float*)d_in[6]; const float* bv = (const float*)d_in[7];
    const float* Wo = (const float*)d_in[8]; const float* bo = (const float*)d_in[9];

    float* out     = (float*)d_out;
    float* weights = out + (size_t)BB * SS * EE;   // tuple: out then weights

    // workspace carve-up (~50 MB)
    char* p = (char*)d_ws;
    _Float16* x_h  = (_Float16*)p; p += (size_t)BB * SS * EE * 2;
    _Float16* Wq_h = (_Float16*)p; p += (size_t)EE * EE * 2;
    _Float16* Wk_h = (_Float16*)p; p += (size_t)EE * EE * 2;
    _Float16* Wv_h = (_Float16*)p; p += (size_t)EE * EE * 2;
    _Float16* Wo_h = (_Float16*)p; p += (size_t)EE * EE * 2;
    _Float16* q_h  = (_Float16*)p; p += (size_t)BB * HH * SS * DK * 2;
    _Float16* k_h  = (_Float16*)p; p += (size_t)BB * HH * SS * DK * 2;
    _Float16* v_h  = (_Float16*)p; p += (size_t)BB * HH * SS * DK * 2;
    _Float16* a_h  = (_Float16*)p; p += (size_t)BB * SS * EE * 2;
    float*    stats = (float*)p;   p += (size_t)BB * HH * SS * 2 * 4;

    const int M = BB * SS, N = EE, K = EE;

    // 1) converts
    {
        int n4 = (BB * SS * EE) / 4;
        cvt_f32_to_f16_kernel<<<n4 / 256, 256, 0, stream>>>(x, x_h, n4);
        int w4 = (EE * EE) / 4;
        cvt_f32_to_f16_kernel<<<w4 / 256, 256, 0, stream>>>(Wq, Wq_h, w4);
        cvt_f32_to_f16_kernel<<<w4 / 256, 256, 0, stream>>>(Wk, Wk_h, w4);
        cvt_f32_to_f16_kernel<<<w4 / 256, 256, 0, stream>>>(Wv, Wv_h, w4);
        cvt_f32_to_f16_kernel<<<w4 / 256, 256, 0, stream>>>(Wo, Wo_h, w4);
    }

    // 2) Q/K/V projections
    dim3 gg(N / 64, M / 128), gb(256);
    gemm_kernel<0><<<gg, gb, 0, stream>>>(x_h, Wq_h, bq, q_h, M, N, K);
    gemm_kernel<0><<<gg, gb, 0, stream>>>(x_h, Wk_h, bk, k_h, M, N, K);
    gemm_kernel<0><<<gg, gb, 0, stream>>>(x_h, Wv_h, bv, v_h, M, N, K);

    // 3) raw scores + online softmax stats
    scores_kernel<<<dim3(SS / 16, BB * HH), 32, 0, stream>>>(q_h, k_h, weights, stats);

    // 4) normalize + mask zeroing
    {
        size_t total4 = ((size_t)BB * HH * SS * SS) / 4;
        norm_mask_kernel<<<(unsigned)(total4 / 256), 256, 0, stream>>>(weights, stats, mask);
    }

    // 5) attn = weights @ V
    attn_kernel<<<dim3(SS / 128, BB * HH), 256, 0, stream>>>(weights, v_h, a_h);

    // 6) out = attn @ Wo + bo
    gemm_kernel<1><<<gg, gb, 0, stream>>>(a_h, Wo_h, bo, out, M, N, K);
}
